// LearnableHarmonicGraph_2894807958200
// MI455X (gfx1250) — matrix operand order repaired and checked by reading, not compiled, not run on hardware
//
#include <hip/hip_runtime.h>
#include <hip/hip_bf16.h>

typedef __attribute__((ext_vector_type(16))) _Float16 v16h;
typedef __attribute__((ext_vector_type(8)))  _Float16 v8h;
typedef __attribute__((ext_vector_type(8)))  float    v8f;

#define WMMA_F16(a, b, c) \
  __builtin_amdgcn_wmma_f32_16x16x32_f16(false, (a), false, (b), (short)0, (c), false, false)

// ---------------------------------------------------------------------------
// Kernel 1: sparse adjacency from band embeddings (31x31, top-5 cosine sim,
// symmetrized). One wave; perf-irrelevant.
// ---------------------------------------------------------------------------
__global__ void adj_kernel(const float* __restrict__ be, float* __restrict__ Aout) {
  __shared__ float ne[31][64];
  __shared__ float Ar[31][31];
  const int lane = threadIdx.x;
  if (lane < 31) {
    float s = 0.f;
    for (int j = 0; j < 64; ++j) { float v = be[lane * 64 + j]; s += v * v; }
    float inv = 1.f / (sqrtf(s) + 1e-8f);
    for (int j = 0; j < 64; ++j) ne[lane][j] = be[lane * 64 + j] * inv;
  }
  __syncthreads();
  if (lane < 31) {
    float sim[31];
    float Arow[31];
    for (int m = 0; m < 31; ++m) {
      float d = 0.f;
      for (int j = 0; j < 64; ++j) d += ne[lane][j] * ne[m][j];
      sim[m] = d;
      Arow[m] = 0.f;
    }
    unsigned picked = 0u;
    for (int t = 0; t < 5; ++t) {
      float best = -1e30f; int bi = 0;
      for (int m = 0; m < 31; ++m)
        if (!((picked >> m) & 1u) && sim[m] > best) { best = sim[m]; bi = m; }
      picked |= (1u << bi);
      Arow[bi] = best;
    }
    for (int m = 0; m < 31; ++m) Ar[lane][m] = Arow[m];
  }
  __syncthreads();
  if (lane < 31)
    for (int m = 0; m < 31; ++m)
      Aout[lane * 31 + m] = 0.5f * (Ar[lane][m] + Ar[m][lane]);
}

// ---------------------------------------------------------------------------
// Fragment loaders per CDNA5 ISA 7.12.2 (wave32, 16-bit 16x16x32)
// A: row m = lane%16; K = (lane>=16 ? 8 : 0) + {0..7, 16..23}
// B (from [n][k]-major buffer): col n = lane%16; K = (lane>=16 ? 8 : 0) + {0..7, 16..23}
// ---------------------------------------------------------------------------
__device__ __forceinline__ v16h load_afrag(const _Float16* base, int ld, int lane) {
  const _Float16* p = base + (lane & 15) * ld + ((lane >> 4) << 3);
  v16h r;
  ((v8h*)&r)[0] = *(const v8h*)(p);
  ((v8h*)&r)[1] = *(const v8h*)(p + 16);
  return r;
}
__device__ __forceinline__ v16h load_bfrag(const _Float16* baseT, int ld, int lane) {
  const _Float16* p = baseT + (lane & 15) * ld + ((lane >> 4) << 3);
  v16h r;
  ((v8h*)&r)[0] = *(const v8h*)(p);
  ((v8h*)&r)[1] = *(const v8h*)(p + 16);
  return r;
}

__device__ __forceinline__ float wred_max(float v) {
#pragma unroll
  for (int off = 16; off > 0; off >>= 1) v = fmaxf(v, __shfl_xor(v, off, 32));
  return v;
}
__device__ __forceinline__ float wred_sum(float v) {
#pragma unroll
  for (int off = 16; off > 0; off >>= 1) v += __shfl_xor(v, off, 32);
  return v;
}

// ---------------------------------------------------------------------------
// Fused: gather/transpose x -> QKV proj -> masked 4-head attention over 31
// nodes -> output proj -> scatter to [B,N,T,K].
// 256 threads = 8 waves; 2 token slots of 4 waves; 16 iterations per block.
// Dynamic LDS: 262 KB.
// ---------------------------------------------------------------------------
#define OFF_WT   0                  // [4][128][128] f16, [mat][n][k] (transposed)
#define OFF_BIAS 131072             // [4][128] f32
#define OFF_ADJ  133120             // [32][32] f32
#define OFF_SLOT 137216             // 2 x 65536
#define SLOT_BYTES 65536
#define SMEM_BYTES (137216 + 2 * SLOT_BYTES)

__global__ void __launch_bounds__(256)
harmonic_kernel(const float* __restrict__ x, const float* __restrict__ adj,
                const float* __restrict__ Wq, const float* __restrict__ bq,
                const float* __restrict__ Wk, const float* __restrict__ bk,
                const float* __restrict__ Wv, const float* __restrict__ bv,
                const float* __restrict__ Wo, const float* __restrict__ bo,
                float* __restrict__ out) {
  extern __shared__ char smem[];
  _Float16* sWt   = (_Float16*)(smem + OFF_WT);
  float*    sBias = (float*)(smem + OFF_BIAS);
  float*    sAdj  = (float*)(smem + OFF_ADJ);

  const int tid  = threadIdx.x;
  const int lane = tid & 31;
  const int wv   = tid >> 5;   // 0..7
  const int slot = wv >> 2;    // 0..1
  const int ws   = wv & 3;     // wave within slot -> head id / n-tile pair

  char* sl = smem + OFF_SLOT + slot * SLOT_BYTES;
  _Float16* sX  = (_Float16*)(sl);           // [32][128]
  _Float16* sQ  = (_Float16*)(sl + 8192);    // [32][128]
  _Float16* sK  = (_Float16*)(sl + 16384);   // [32][128]
  _Float16* sVt = (_Float16*)(sl + 24576);   // [128][32] (feature-major)
  float*    sS  = (float*)(sl + 32768);      // [4][32][32]
  _Float16* sP  = (_Float16*)(sl + 49152);   // [4][32][32]
  _Float16* sO  = (_Float16*)(sl + 57344);   // [32][128]

  // ---- stage weights (transposed to [n][k]), biases, adjacency ----
  const float* Ws[4] = {Wq, Wk, Wv, Wo};
  for (int mat = 0; mat < 4; ++mat) {
    const float* W = Ws[mat];
    for (int idx = tid; idx < 128 * 128; idx += 256) {
      int kk = idx >> 7, nn = idx & 127;
      sWt[mat * 16384 + nn * 128 + kk] = (_Float16)W[idx];
    }
  }
  const float* bs[4] = {bq, bk, bv, bo};
  for (int idx = tid; idx < 512; idx += 256) sBias[idx] = bs[idx >> 7][idx & 127];
  for (int idx = tid; idx < 1024; idx += 256) {
    int m = idx >> 5, n = idx & 31;
    sAdj[idx] = (m < 31 && n < 31) ? adj[m * 31 + n] : 0.f;
  }
  __syncthreads();

  const int n_feat = tid & 127;
  const int lo = lane & 15;
  const int hi8 = (lane >> 4) << 3;

  for (int it = 0; it < 16; ++it) {
    const int bt = blockIdx.x * 32 + it * 2 + slot;  // 0..8191
    const int bb = bt >> 10;
    const int tt = bt & 1023;

    // ---- gather x[b, n, t, 0..30] -> sX[k][n] (f16), pad row 31 with 0 ----
    {
      const float* xp = x + (((bb * 128 + n_feat) * 1024) + tt) * 31;
#pragma unroll
      for (int k = 0; k < 31; ++k) sX[k * 128 + n_feat] = (_Float16)xp[k];
      sX[31 * 128 + n_feat] = (_Float16)0.f;
    }
    // ---- prefetch next iteration's x lines into L2 (overlaps compute) ----
    if (it < 15) {
      const int bt2 = bt + 2;
      const int b2 = bt2 >> 10, t2 = bt2 & 1023;
      const float* pn = x + (((b2 * 128 + n_feat) * 1024) + t2) * 31;
      __builtin_prefetch(pn, 0, 1);
      __builtin_prefetch(pn + 16, 0, 1);   // 31 floats may span 2-3 cachelines
    }
    __syncthreads();

    // ---- Q, K, V projections: [32,128] @ [128,128] ----
    {
      // X A-fragments: load once, reuse across Q/K/V
      v16h ax[2][4];
#pragma unroll
      for (int mt = 0; mt < 2; ++mt)
#pragma unroll
        for (int kb = 0; kb < 4; ++kb)
          ax[mt][kb] = load_afrag(sX + mt * 2048 + kb * 32, 128, lane);

#pragma unroll
      for (int mat = 0; mat < 3; ++mat) {
#pragma unroll
        for (int nt2 = 0; nt2 < 2; ++nt2) {
          const int nt = ws * 2 + nt2;
          const _Float16* wb = sWt + mat * 16384 + nt * 2048;
          // B-fragments: load once, reuse across both M-tiles
          v16h b0 = load_bfrag(wb + 0,  128, lane);
          v16h b1 = load_bfrag(wb + 32, 128, lane);
          v16h b2 = load_bfrag(wb + 64, 128, lane);
          v16h b3 = load_bfrag(wb + 96, 128, lane);
          const int colb = nt * 16 + lo;
          const float bias = sBias[mat * 128 + colb];
#pragma unroll
          for (int mt = 0; mt < 2; ++mt) {
            v8f acc = {};
            acc = WMMA_F16(ax[mt][0], b0, acc);
            acc = WMMA_F16(ax[mt][1], b1, acc);
            acc = WMMA_F16(ax[mt][2], b2, acc);
            acc = WMMA_F16(ax[mt][3], b3, acc);
            const int rowb = mt * 16 + hi8;
#pragma unroll
            for (int r = 0; r < 8; ++r) {
              float val = acc[r] + bias;
              int row = rowb + r;
              if (mat == 0)      sQ[row * 128 + colb] = (_Float16)val;
              else if (mat == 1) sK[row * 128 + colb] = (_Float16)val;
              else               sVt[colb * 32 + row] = (_Float16)val;  // transposed
            }
          }
        }
      }
    }
    __syncthreads();

    // ---- attention, head h = ws (data is wave-private by construction) ----
    {
      const int h = ws;
      // scores = q_h @ k_h^T  (K = hd = 32, single WMMA step)
      {
        v16h aq0 = load_afrag(sQ + 0    + h * 32, 128, lane);
        v16h aq1 = load_afrag(sQ + 2048 + h * 32, 128, lane);
        v16h bk0 = load_bfrag(sK + 0    + h * 32, 128, lane);
        v16h bk1 = load_bfrag(sK + 2048 + h * 32, 128, lane);
#pragma unroll
        for (int mt = 0; mt < 2; ++mt) {
#pragma unroll
          for (int nt = 0; nt < 2; ++nt) {
            v8f acc = {};
            acc = WMMA_F16(mt ? aq1 : aq0, nt ? bk1 : bk0, acc);
            const int colb = nt * 16 + lo;
            const int rowb = mt * 16 + hi8;
#pragma unroll
            for (int r = 0; r < 8; ++r) sS[h * 1024 + (rowb + r) * 32 + colb] = acc[r];
          }
        }
      }
      // masked softmax over 31 columns; lane = column
      const float scale = 0.17677669529663687f;  // 1/sqrt(32)
      for (int m = 0; m < 31; ++m) {
        const bool valid = (lane < 31) && (sAdj[m * 32 + lane] != 0.f);
        float v = valid ? sS[h * 1024 + m * 32 + lane] * scale : -__builtin_inff();
        float mx = wred_max(v);
        float e = valid ? __expf(v - mx) : 0.f;
        float sum = wred_sum(e);
        float p = (sum > 0.f) ? e / sum : 0.f;  // nan_to_num for all-masked rows
        sP[h * 1024 + m * 32 + lane] = (_Float16)p;
      }
      sP[h * 1024 + 31 * 32 + lane] = (_Float16)0.f;  // zero padded row

      // out_h = P @ v_h  (K = 31 padded to 32; P pad col/row are zero)
      {
        v16h ap0 = load_afrag(sP + h * 1024 + 0,   32, lane);
        v16h ap1 = load_afrag(sP + h * 1024 + 512, 32, lane);
        v16h bv0 = load_bfrag(sVt + (h * 32 + 0)  * 32, 32, lane);
        v16h bv1 = load_bfrag(sVt + (h * 32 + 16) * 32, 32, lane);
#pragma unroll
        for (int mt = 0; mt < 2; ++mt) {
#pragma unroll
          for (int nt = 0; nt < 2; ++nt) {
            v8f acc = {};
            acc = WMMA_F16(mt ? ap1 : ap0, nt ? bv1 : bv0, acc);
            const int colb = h * 32 + nt * 16 + lo;
            const int rowb = mt * 16 + hi8;
#pragma unroll
            for (int r = 0; r < 8; ++r) sO[(rowb + r) * 128 + colb] = (_Float16)acc[r];
          }
        }
      }
    }
    __syncthreads();  // sO columns come from all 4 waves of the slot

    // ---- output projection + scatter store to [B, N, T, K] ----
    {
      v16h ao[2][4];
#pragma unroll
      for (int mt = 0; mt < 2; ++mt)
#pragma unroll
        for (int kb = 0; kb < 4; ++kb)
          ao[mt][kb] = load_afrag(sO + mt * 2048 + kb * 32, 128, lane);

#pragma unroll
      for (int nt2 = 0; nt2 < 2; ++nt2) {
        const int nt = ws * 2 + nt2;
        const _Float16* wb = sWt + 3 * 16384 + nt * 2048;
        v16h b0 = load_bfrag(wb + 0,  128, lane);
        v16h b1 = load_bfrag(wb + 32, 128, lane);
        v16h b2 = load_bfrag(wb + 64, 128, lane);
        v16h b3 = load_bfrag(wb + 96, 128, lane);
        const int colb = nt * 16 + lo;
        const float bias = sBias[3 * 128 + colb];
        float* op = out + (((bb * 128 + colb) * 1024) + tt) * 31;
#pragma unroll
        for (int mt = 0; mt < 2; ++mt) {
          v8f acc = {};
          acc = WMMA_F16(ao[mt][0], b0, acc);
          acc = WMMA_F16(ao[mt][1], b1, acc);
          acc = WMMA_F16(ao[mt][2], b2, acc);
          acc = WMMA_F16(ao[mt][3], b3, acc);
          const int rowb = mt * 16 + hi8;
#pragma unroll
          for (int r = 0; r < 8; ++r) {
            int krow = rowb + r;
            if (krow < 31) op[krow] = acc[r] + bias;
          }
        }
      }
    }
    __syncthreads();  // before sX/sO are overwritten next iteration
  }
}

// ---------------------------------------------------------------------------
extern "C" void kernel_launch(void* const* d_in, const int* in_sizes, int n_in,
                              void* d_out, int out_size, void* d_ws, size_t ws_size,
                              hipStream_t stream) {
  (void)in_sizes; (void)n_in; (void)out_size; (void)ws_size;
  const float* x  = (const float*)d_in[0];
  const float* be = (const float*)d_in[1];
  const float* Wq = (const float*)d_in[2];
  const float* bq = (const float*)d_in[3];
  const float* Wk = (const float*)d_in[4];
  const float* bk = (const float*)d_in[5];
  const float* Wv = (const float*)d_in[6];
  const float* bv = (const float*)d_in[7];
  const float* Wo = (const float*)d_in[8];
  const float* bo = (const float*)d_in[9];
  float* out = (float*)d_out;
  float* dA  = (float*)d_ws;  // 31*31 floats

  adj_kernel<<<1, 32, 0, stream>>>(be, dA);

  hipFuncSetAttribute(reinterpret_cast<const void*>(harmonic_kernel),
                      hipFuncAttributeMaxDynamicSharedMemorySize, SMEM_BYTES);
  harmonic_kernel<<<256, 256, SMEM_BYTES, stream>>>(
      x, dA, Wq, bq, Wk, bk, Wv, bv, Wo, bo, out);
}
